// GlobalXCALayer_9509057593878
// MI455X (gfx1250) — matrix operand order repaired and checked by reading, not compiled, not run on hardware
//
#include <hip/hip_runtime.h>

// ---------------------------------------------------------------------------
// GlobalXCALayer on MI455X (gfx1250, wave32, WMMA)
//   x:[1,64,64,64,96] f32.  LN -> conv3d(96->288,k3) as implicit GEMM (WMMA
//   f16 in / f32 acc, async global->LDS staging) -> region channel-attention
//   (24x24 per f,head) -> 96x96 projection (WMMA) + residual.
// ---------------------------------------------------------------------------

typedef __attribute__((ext_vector_type(16))) _Float16 v16h;
typedef __attribute__((ext_vector_type(8)))  _Float16 v8h;
typedef __attribute__((ext_vector_type(8)))  float    v8f;
typedef __attribute__((ext_vector_type(4)))  int      v4i;

union ABu { v16h v; v8h h[2]; };

__device__ __forceinline__ v8f wmma_f16(const ABu& a, const ABu& b, v8f c) {
  return __builtin_amdgcn_wmma_f32_16x16x32_f16(
      /*neg_a=*/false, a.v, /*neg_b=*/false, b.v,
      /*c_mod=*/(short)0, c, /*reuse_a=*/false, /*reuse_b=*/false);
}

#if __has_builtin(__builtin_amdgcn_global_load_async_to_lds_b128) && \
    __has_builtin(__builtin_amdgcn_s_wait_asynccnt)
#define HAVE_ASYNC_LDS 1
#else
#define HAVE_ASYNC_LDS 0
#endif

#if HAVE_ASYNC_LDS
typedef __attribute__((address_space(1))) v4i* as1_v4i_p;   // global
typedef __attribute__((address_space(3))) v4i* as3_v4i_p;   // LDS
// 16B global -> LDS async copy (ASYNCcnt tracked, no VGPR round trip)
__device__ __forceinline__ void async_copy16(const _Float16* g, _Float16* l) {
  __builtin_amdgcn_global_load_async_to_lds_b128(
      (as1_v4i_p)(g), (as3_v4i_p)(l), 0, 0);
}
#endif

#define NVOX   262144      // 64*64*64
#define C_IN   96
#define C_QKV  288
#define HEADS  4
#define CH     24          // C_IN / HEADS
#define NF     64          // 4*4*4 intra-region positions
#define NG     4096        // 16^3 regions

// ---------------------------------------------------------------------------
// K1: LayerNorm over channels, emit f16.  One wave per voxel, lane handles
//     channels {lane, lane+32, lane+64}; wave32 shuffle reduction.
// ---------------------------------------------------------------------------
__global__ __launch_bounds__(256) void ln_kernel(
    const float* __restrict__ x, const float* __restrict__ gamma,
    const float* __restrict__ beta, _Float16* __restrict__ xn) {
  const int lane = threadIdx.x & 31;
  const int wave = threadIdx.x >> 5;
  const size_t v = (size_t)blockIdx.x * 8 + wave;
  const float* xv = x + v * C_IN;
  float x0 = xv[lane], x1 = xv[lane + 32], x2 = xv[lane + 64];
  float s  = x0 + x1 + x2;
  float s2 = x0 * x0 + x1 * x1 + x2 * x2;
  #pragma unroll
  for (int m = 16; m >= 1; m >>= 1) {
    s  += __shfl_xor(s,  m, 32);
    s2 += __shfl_xor(s2, m, 32);
  }
  const float mean = s * (1.0f / 96.0f);
  const float var  = s2 * (1.0f / 96.0f) - mean * mean;
  const float inv  = rsqrtf(var + 1e-5f);
  _Float16* o = xn + v * C_IN;
  o[lane]      = (_Float16)((x0 - mean) * inv * gamma[lane]      + beta[lane]);
  o[lane + 32] = (_Float16)((x1 - mean) * inv * gamma[lane + 32] + beta[lane + 32]);
  o[lane + 64] = (_Float16)((x2 - mean) * inv * gamma[lane + 64] + beta[lane + 64]);
}

// ---------------------------------------------------------------------------
// K2: repack weights to f16 in WMMA B-fragment-friendly layout:
//   wB[((tap*3+kc)*288 + n)*32 + k] = qkv_w[tap][kc*32+k][n]
//   wO[(kc*96 + n)*32 + k]          = out_w[n][kc*32+k]
// so each lane's 16-element K-run is two contiguous 16B loads.
// ---------------------------------------------------------------------------
__global__ __launch_bounds__(256) void cvt_weights(
    const float* __restrict__ qkv_w, const float* __restrict__ out_w,
    _Float16* __restrict__ wB, _Float16* __restrict__ wO) {
  const int idx = blockIdx.x * blockDim.x + threadIdx.x;
  if (idx < 27 * 96 * 288) {
    int co  = idx % 288;
    int rest = idx / 288;
    int ci  = rest % 96;
    int tap = rest / 96;
    int kc = ci >> 5, k = ci & 31;
    wB[((size_t)(tap * 3 + kc) * 288 + co) * 32 + k] = (_Float16)qkv_w[idx];
  } else if (idx < 27 * 96 * 288 + 96 * 96) {
    int j = idx - 27 * 96 * 288;
    int ci = j % 96, n = j / 96;
    int kc = ci >> 5, k = ci & 31;
    wO[((size_t)(kc * 96 + n)) * 32 + k] = (_Float16)out_w[j];
  }
}

// ---------------------------------------------------------------------------
// K3: conv3d as implicit GEMM.  M = voxels (tile 32), N = 288 (grid.y * 96,
//     3 waves x two 16-col subtiles), K = 27 taps x 96 ch (3 x 32 chunks).
//     A 34-row halo strip (rows d0-1..d0+32, 96 ch) staged ONCE per (dh,dw)
//     via async global->LDS; the dd taps read it at +dd row offset.  Each
//     wave: 2 A frags x 2 B frags -> 4 WMMAs per (tap,kc).
// ---------------------------------------------------------------------------
__global__ __launch_bounds__(96) void conv_qkv(
    const _Float16* __restrict__ xn, const _Float16* __restrict__ wB,
    const float* __restrict__ qkv_b, _Float16* __restrict__ qkv) {
  __shared__ alignas(16) _Float16 As[34][96];   // storage row i -> d = d0+i-1

  const int m0   = blockIdx.x * 32;
  const int n0   = blockIdx.y * 96;
  const int tid  = threadIdx.x;
  const int lane = tid & 31;
  const int wave = tid >> 5;

  const int h0 = m0 >> 12;
  const int w0 = (m0 >> 6) & 63;
  const int d0 = m0 & 63;          // 32 consecutive d (same h0,w0)

  v8f acc00 = {}, acc01 = {}, acc10 = {}, acc11 = {};
  const int Nn0  = n0 + wave * 32 + (lane & 15);   // subtile0 column
  const int kb2  = (lane < 16) ? 0 : 16;           // B frag K base
  const int arow = lane & 15;                      // A frag M (within 16)
  const int akb  = (lane < 16) ? 0 : 16;           // A frag K base

  for (int dhw = 0; dhw < 9; ++dhw) {
    const int dh = dhw / 3, dw = dhw % 3;
    const int hh = h0 + dh - 1, ww = w0 + dw - 1;
    const bool hwok = ((unsigned)hh < 64u) && ((unsigned)ww < 64u);
    const _Float16* grow =
        xn + (((size_t)(hh * 64 + ww) * 64 + d0 - 1)) * C_IN;  // row i: +i*96

    // stage 34 rows x 96 ch as 16B chunks (12 per row), zero-padded OOB
    for (int ch = tid; ch < 34 * 12; ch += 96) {
      const int r  = ch / 12;
      const int cc = ch - r * 12;
      const int dn = d0 + r - 1;
      _Float16* lp = &As[r][cc * 8];
      const bool ok = hwok && ((unsigned)dn < 64u);
#if HAVE_ASYNC_LDS
      if (ok) {
        async_copy16(grow + (size_t)r * C_IN + cc * 8, lp);
      } else {
        v8h z = {};
        *(v8h*)lp = z;
      }
#else
      v8h z = {};
      *(v8h*)lp = ok ? *(const v8h*)(grow + (size_t)r * C_IN + cc * 8) : z;
#endif
    }
#if HAVE_ASYNC_LDS
    __builtin_amdgcn_s_wait_asynccnt(0);
#endif
    __syncthreads();

    for (int dd = 0; dd < 3; ++dd) {
      const int tap = dhw * 3 + dd;
      #pragma unroll
      for (int kc = 0; kc < 3; ++kc) {
        ABu a0, a1;
        a0.h[0] = *(const v8h*)&As[arow + dd][kc * 32 + akb];
        a0.h[1] = *(const v8h*)&As[arow + dd][kc * 32 + akb + 8];
        a1.h[0] = *(const v8h*)&As[arow + dd + 16][kc * 32 + akb];
        a1.h[1] = *(const v8h*)&As[arow + dd + 16][kc * 32 + akb + 8];

        const _Float16* bp = wB + (((size_t)(tap * 3 + kc) * 288 + Nn0) * 32 + kb2);
        ABu b0; b0.h[0] = *(const v8h*)bp;       b0.h[1] = *(const v8h*)(bp + 8);
        const _Float16* bq = bp + 16 * 32;       // subtile1: +16 columns
        ABu b1; b1.h[0] = *(const v8h*)bq;       b1.h[1] = *(const v8h*)(bq + 8);

        acc00 = wmma_f16(a0, b0, acc00);
        acc01 = wmma_f16(a0, b1, acc01);
        acc10 = wmma_f16(a1, b0, acc10);
        acc11 = wmma_f16(a1, b1, acc11);
      }
    }
    __syncthreads();
  }

  // C/D layout: lane N = L%16, VGPR j -> M = j + 8*(L>=16)
  const int nA = Nn0, nB = Nn0 + 16;
  const float bA = qkv_b[nA], bB = qkv_b[nB];
  const int mbase = m0 + ((lane < 16) ? 0 : 8);
  #pragma unroll
  for (int j = 0; j < 8; ++j) {
    const size_t mA = (size_t)(mbase + j);
    const size_t mB = mA + 16;
    qkv[mA * C_QKV + nA] = (_Float16)(acc00[j] + bA);
    qkv[mA * C_QKV + nB] = (_Float16)(acc01[j] + bB);
    qkv[mB * C_QKV + nA] = (_Float16)(acc10[j] + bA);
    qkv[mB * C_QKV + nB] = (_Float16)(acc11[j] + bB);
  }
}

// ---------------------------------------------------------------------------
// K4: inverse L2 norms of q,k over the region axis (4096) per (f, channel).
//     qkv columns 0..95 = q, 96..191 = k, so thread c in [0,192) covers both.
// ---------------------------------------------------------------------------
__global__ __launch_bounds__(192) void norm_kernel(
    const _Float16* __restrict__ qkv, float* __restrict__ qkinv) {
  const int f = blockIdx.x;
  const int c = threadIdx.x;            // 0..191
  const int fh = f >> 4, fw = (f >> 2) & 3, fd = f & 3;
  float s = 0.0f;
  for (int g = 0; g < NG; ++g) {
    const int gh = g >> 8, gw = (g >> 4) & 15, gd = g & 15;
    const size_t v = ((size_t)((gh * 4 + fh) * 64 + (gw * 4 + fw))) * 64 + (gd * 4 + fd);
    const float q = (float)qkv[v * C_QKV + c];
    s = fmaf(q, q, s);
  }
  qkinv[f * 192 + c] = 1.0f / fmaxf(sqrtf(s), 1e-12f);
}

// ---------------------------------------------------------------------------
// K5: channel attention per (f, head): S = qn . kn^T (24x24 over G=4096,
//     LDS-tiled in chunks of 128), softmax over d, out = P . v.
// ---------------------------------------------------------------------------
__global__ __launch_bounds__(256) void attn_kernel(
    const _Float16* __restrict__ qkv, const float* __restrict__ qkinv,
    const float* __restrict__ temperature, _Float16* __restrict__ ao) {
  __shared__ float qt[CH][128];
  __shared__ float kt[CH][128];
  __shared__ float S[CH * CH];

  const int f    = blockIdx.x >> 2;
  const int head = blockIdx.x & 3;
  const int tid  = threadIdx.x;
  const int fh = f >> 4, fw = (f >> 2) & 3, fd = f & 3;
  const int qcol = head * CH;
  const int kcol = 96 + head * CH;
  const int vcol = 192 + head * CH;
  const float* qiv = qkinv + f * 192 + head * CH;
  const float* kiv = qkinv + f * 192 + 96 + head * CH;

  int pl[3]; int np = 0;
  for (int p = tid; p < CH * CH; p += 256) pl[np++] = p;
  float acc[3] = {0.f, 0.f, 0.f};

  for (int chg = 0; chg < NG; chg += 128) {
    for (int e = tid; e < CH * 128; e += 256) {
      const int c = e % CH, gg = e / CH;
      const int g = chg + gg;
      const int gh = g >> 8, gw = (g >> 4) & 15, gd = g & 15;
      const size_t v = ((size_t)((gh * 4 + fh) * 64 + (gw * 4 + fw))) * 64 + (gd * 4 + fd);
      qt[c][gg] = (float)qkv[v * C_QKV + qcol + c] * qiv[c];
      kt[c][gg] = (float)qkv[v * C_QKV + kcol + c] * kiv[c];
    }
    __syncthreads();
    for (int i = 0; i < np; ++i) {
      const int p = pl[i];
      const int c = p / CH, d = p % CH;
      float t = 0.0f;
      for (int gg = 0; gg < 128; ++gg) t = fmaf(qt[c][gg], kt[d][gg], t);
      acc[i] += t;
    }
    __syncthreads();
  }

  const float temp = temperature[head];
  for (int i = 0; i < np; ++i) S[pl[i]] = acc[i] * temp;
  __syncthreads();

  if (tid < CH) {                         // softmax over d, one row per thread
    float m = -1e30f;
    for (int d = 0; d < CH; ++d) m = fmaxf(m, S[tid * CH + d]);
    float sum = 0.0f;
    for (int d = 0; d < CH; ++d) {
      const float e = __expf(S[tid * CH + d] - m);
      S[tid * CH + d] = e;
      sum += e;
    }
    const float r = 1.0f / sum;
    for (int d = 0; d < CH; ++d) S[tid * CH + d] *= r;
  }
  __syncthreads();

  for (int idx = tid; idx < CH * NG; idx += 256) {
    const int c = idx % CH;
    const int g = idx / CH;
    const int gh = g >> 8, gw = (g >> 4) & 15, gd = g & 15;
    const size_t v = ((size_t)((gh * 4 + fh) * 64 + (gw * 4 + fw))) * 64 + (gd * 4 + fd);
    const _Float16* vp = qkv + v * C_QKV + vcol;
    float o = 0.0f;
    for (int d = 0; d < CH; ++d) o = fmaf(S[c * CH + d], (float)vp[d], o);
    ao[v * C_IN + head * CH + c] = (_Float16)o;
  }
}

// ---------------------------------------------------------------------------
// K6: output projection (M=voxels tile 32, N=96, K=96) via WMMA + bias +
//     residual.  Same async LDS staging and 2x2 fragment blocking.
// ---------------------------------------------------------------------------
__global__ __launch_bounds__(96) void proj_kernel(
    const _Float16* __restrict__ ao, const _Float16* __restrict__ wO,
    const float* __restrict__ out_b, const float* __restrict__ x,
    float* __restrict__ out) {
  __shared__ alignas(16) _Float16 As[32][96];
  const int m0   = blockIdx.x * 32;
  const int tid  = threadIdx.x;
  const int lane = tid & 31;
  const int wave = tid >> 5;

  v8f acc00 = {}, acc01 = {}, acc10 = {}, acc11 = {};
  const int Nn0  = wave * 32 + (lane & 15);
  const int kb2  = (lane < 16) ? 0 : 16;
  const int arow = lane & 15;
  const int akb  = (lane < 16) ? 0 : 16;

  for (int ch = tid; ch < 32 * 12; ch += 96) {
    const int r  = ch / 12;
    const int cc = ch - r * 12;
    const _Float16* gp = ao + (size_t)(m0 + r) * C_IN + cc * 8;
    _Float16* lp = &As[r][cc * 8];
#if HAVE_ASYNC_LDS
    async_copy16(gp, lp);
#else
    *(v8h*)lp = *(const v8h*)gp;
#endif
  }
#if HAVE_ASYNC_LDS
  __builtin_amdgcn_s_wait_asynccnt(0);
#endif
  __syncthreads();

  #pragma unroll
  for (int kc = 0; kc < 3; ++kc) {
    ABu a0, a1;
    a0.h[0] = *(const v8h*)&As[arow][kc * 32 + akb];
    a0.h[1] = *(const v8h*)&As[arow][kc * 32 + akb + 8];
    a1.h[0] = *(const v8h*)&As[arow + 16][kc * 32 + akb];
    a1.h[1] = *(const v8h*)&As[arow + 16][kc * 32 + akb + 8];

    const _Float16* bp = wO + ((size_t)(kc * 96 + Nn0) * 32 + kb2);
    ABu b0; b0.h[0] = *(const v8h*)bp;  b0.h[1] = *(const v8h*)(bp + 8);
    const _Float16* bq = bp + 16 * 32;
    ABu b1; b1.h[0] = *(const v8h*)bq;  b1.h[1] = *(const v8h*)(bq + 8);

    acc00 = wmma_f16(a0, b0, acc00);
    acc01 = wmma_f16(a0, b1, acc01);
    acc10 = wmma_f16(a1, b0, acc10);
    acc11 = wmma_f16(a1, b1, acc11);
  }

  const int nA = Nn0, nB = Nn0 + 16;
  const float bA = out_b[nA], bB = out_b[nB];
  const int mbase = m0 + ((lane < 16) ? 0 : 8);
  #pragma unroll
  for (int j = 0; j < 8; ++j) {
    const size_t mA = (size_t)(mbase + j);
    const size_t mB = mA + 16;
    out[mA * C_IN + nA] = acc00[j] + bA + x[mA * C_IN + nA];
    out[mA * C_IN + nB] = acc01[j] + bB + x[mA * C_IN + nB];
    out[mB * C_IN + nA] = acc10[j] + bA + x[mB * C_IN + nA];
    out[mB * C_IN + nB] = acc11[j] + bB + x[mB * C_IN + nB];
  }
}

// ---------------------------------------------------------------------------
extern "C" void kernel_launch(void* const* d_in, const int* in_sizes, int n_in,
                              void* d_out, int out_size, void* d_ws, size_t ws_size,
                              hipStream_t stream) {
  const float* x      = (const float*)d_in[0];
  const float* ln_g   = (const float*)d_in[1];
  const float* ln_b   = (const float*)d_in[2];
  const float* qkv_w  = (const float*)d_in[3];
  const float* qkv_b  = (const float*)d_in[4];
  const float* out_w  = (const float*)d_in[5];
  const float* out_b  = (const float*)d_in[6];
  const float* temp   = (const float*)d_in[7];
  float* out = (float*)d_out;

  char* ws = (char*)d_ws;
  size_t off = 0;
  _Float16* xnf  = (_Float16*)(ws + off); off += (size_t)NVOX * C_IN  * 2;  // 50 MB
  _Float16* qkvf = (_Float16*)(ws + off); off += (size_t)NVOX * C_QKV * 2;  // 151 MB
  _Float16* aof  = (_Float16*)(ws + off); off += (size_t)NVOX * C_IN  * 2;  // 50 MB
  _Float16* wB   = (_Float16*)(ws + off); off += (size_t)27 * 96 * 288 * 2;
  off = (off + 255) & ~(size_t)255;
  _Float16* wO   = (_Float16*)(ws + off); off += (size_t)96 * 96 * 2;
  off = (off + 255) & ~(size_t)255;
  float* qkinv   = (float*)(ws + off);    off += (size_t)NF * 192 * 4;

  ln_kernel  <<<NVOX / 8, 256, 0, stream>>>(x, ln_g, ln_b, xnf);
  cvt_weights<<<(27 * 96 * 288 + 96 * 96 + 255) / 256, 256, 0, stream>>>(qkv_w, out_w, wB, wO);
  conv_qkv   <<<dim3(NVOX / 32, 3), 96, 0, stream>>>(xnf, wB, qkv_b, qkvf);
  norm_kernel<<<NF, 192, 0, stream>>>(qkvf, qkinv);
  attn_kernel<<<NF * HEADS, 256, 0, stream>>>(qkvf, qkinv, temp, aof);
  proj_kernel<<<NVOX / 32, 96, 0, stream>>>(aof, wO, out_b, x, out);
}
